// PredNet_8598524527139
// MI455X (gfx1250) — compile-verified, matrix operand order checked by reading, not run on hardware
//
#include <hip/hip_runtime.h>
#include <math.h>
#include <stdint.h>

typedef __attribute__((ext_vector_type(16))) _Float16 v16h;
typedef __attribute__((ext_vector_type(8)))  float    v8f;

#define BN_EPS 1e-3f

// =====================================================================
// Implicit-GEMM 3x3 SAME conv (stride 1), NHWC, f16 WMMA, f32 accumulate
//   out[M x Cout] (+)= im2col(in)[M x 9*Cin] * wgtT^T (+bias)
// wgtT: pre-transposed, padded f16 weights [Npad][Kpad] (N-major, K contig)
// Block: 256 threads = 8 waves. Tile: 128 pixels x 64 out-channels.
// B tile staged via GLOBAL_LOAD_ASYNC_TO_LDS_B128 (ASYNCcnt).
// =====================================================================
__global__ __launch_bounds__(256)
void k_conv3x3_wmma(float* __restrict__ out,
                    const float* __restrict__ in,
                    const _Float16* __restrict__ wgtT,  // [Npad][Kpad] f16
                    const float* __restrict__ bias,     // [Cout] or null
                    int Bn, int H, int W, int Cin, int Cout, int accumulate)
{
    __shared__ __align__(32) _Float16 sA[128 * 32];
    __shared__ __align__(32) _Float16 sB[64 * 32];

    const int tid   = threadIdx.x;
    const int wave  = tid >> 5;
    const int lane  = tid & 31;
    const int l16   = lane & 15;
    const int khalf = (lane >> 4) * 16;   // K-half of the fragment
    const int mbase = blockIdx.x * 128;
    const int nbase = blockIdx.y * 64;
    const int HW    = H * W;
    const int M     = Bn * HW;
    const int Ktot  = 9 * Cin;
    const int Kpad  = (Ktot + 31) & ~31;

    // A staging: 2 threads per tile row (128 rows), 16 K-elems each
    const int am = tid >> 1;
    const int ak = (tid & 1) * 16;
    const int gm = mbase + am;
    int b0 = 0, y0 = 0, x0 = 0;
    const bool mv = gm < M;
    if (mv) { b0 = gm / HW; int r = gm - b0 * HW; y0 = r / W; x0 = r - y0 * W; }

    // B staging: 4 threads per weight row (64 rows), 16 bytes (8 halves) each
    const int bn = tid >> 2;
    const int bq = (tid & 3) * 8;
    const unsigned ldsb = (unsigned)(uintptr_t)&sB[bn * 32 + bq];   // LDS byte offset in addr[31:0]
    const _Float16* gbrow = wgtT + (size_t)(nbase + bn) * Kpad + bq;

    v8f acc[4];
    const v8f vzero = {};
#pragma unroll
    for (int j = 0; j < 4; ++j) acc[j] = vzero;

    for (int k0 = 0; k0 < Ktot; k0 += 32) {
        // ---- async-stage B tile: global -> LDS, 16B per lane (4KB total) ----
        {
            unsigned long long ga = (unsigned long long)(uintptr_t)(gbrow + k0);
            asm volatile("global_load_async_to_lds_b128 %0, %1, off"
                         :: "v"(ldsb), "v"(ga) : "memory");
        }
        if (tid == 0 && k0 + 32 < Ktot)   // global_prefetch_b8: next weight chunk
            __builtin_prefetch(wgtT + (size_t)nbase * Kpad + k0 + 32, 0, 0);

        // ---- stage A tile (im2col gather, zero-pad SAME borders) ----
        {
            int k   = k0 + ak;
            int tap = k / Cin;
            int ci  = k - tap * Cin;
#pragma unroll 4
            for (int i = 0; i < 16; ++i) {
                float v = 0.f;
                if (mv && (k + i) < Ktot) {
                    int ky = tap / 3;
                    int kx = tap - ky * 3;
                    int yy = y0 + ky - 1, xx = x0 + kx - 1;
                    if ((unsigned)yy < (unsigned)H && (unsigned)xx < (unsigned)W)
                        v = in[(size_t)((b0 * H + yy) * W + xx) * Cin + ci];
                }
                sA[am * 32 + ak + i] = (_Float16)v;
                if (++ci == Cin) { ci = 0; ++tap; }
            }
        }
        asm volatile("s_wait_asynccnt 0" ::: "memory");
        __syncthreads();

        // A frag: lane%16 = row in wave's 16-row strip, lane/16 = K half
        v16h afrag = *(const v16h*)&sA[(wave * 16 + l16) * 32 + khalf];
#pragma unroll
        for (int j = 0; j < 4; ++j) {
            // B frag: lane%16 = column, lane/16 = K half
            v16h bfrag = *(const v16h*)&sB[(j * 16 + l16) * 32 + khalf];
            acc[j] = __builtin_amdgcn_wmma_f32_16x16x32_f16(
                false, afrag, false, bfrag, (short)0, acc[j], false, false);
        }
        __syncthreads();
    }

    // C/D layout: VGPR r, lanes 0-15 -> M=r, lanes 16-31 -> M=8+r; N = lane%16
    const int mrow = mbase + wave * 16 + (lane >> 4) * 8;
#pragma unroll
    for (int j = 0; j < 4; ++j) {
        int n = nbase + j * 16 + l16;
        if (n >= Cout) continue;
        float bv = bias ? bias[n] : 0.f;
#pragma unroll
        for (int r = 0; r < 8; ++r) {
            int m = mrow + r;
            if (m < M) {
                size_t o = (size_t)m * Cout + n;
                if (accumulate) out[o] += acc[j][r];
                else            out[o]  = acc[j][r] + bv;
            }
        }
    }
}

// =====================================================================
// Weight prep: fp32 HWIO [Ktot][Cout] -> f16 transposed padded [Npad][Kpad]
// =====================================================================
__global__ void k_wtr(_Float16* d, const float* s, int Cout, int Ktot, int Kpad, int n) {
    int i = blockIdx.x * blockDim.x + threadIdx.x;
    if (i >= n) return;
    int k  = i % Kpad;
    int nn = i / Kpad;
    _Float16 v = (_Float16)0.f;
    if (nn < Cout && k < Ktot) v = (_Float16)s[(size_t)k * Cout + nn];
    d[i] = v;
}

// =====================================================================
// Elementwise helpers
// =====================================================================
__global__ void k_zero(float* d, int n) {
    int i = blockIdx.x * blockDim.x + threadIdx.x;
    if (i < n) d[i] = 0.f;
}

// dst[..., off+c] = src[..., c]   (channel concat copy)
__global__ void k_copy_off(float* dst, const float* src, int Cs, int Cd, int off, int n) {
    int i = blockIdx.x * blockDim.x + threadIdx.x;
    if (i >= n) return;
    int c = i % Cs, p = i / Cs;
    dst[(size_t)p * Cd + off + c] = src[i];
}

// dst(B,H,W,Cd)[..., off+c] = src(B,H/2,W/2,Cs) nearest-2x upsample
__global__ void k_upsample_off(float* dst, const float* src,
                               int H, int W, int Cs, int Cd, int off, int n) {
    int i = blockIdx.x * blockDim.x + threadIdx.x;
    if (i >= n) return;
    int c = i % Cs, p = i / Cs;
    int x = p % W; p /= W;
    int y = p % H; int b = p / H;
    size_t si = ((size_t)(b * (H / 2) + (y >> 1)) * (W / 2) + (x >> 1)) * Cs + c;
    dst[((size_t)(b * H + y) * W + x) * Cd + off + c] = src[si];
}

__device__ __forceinline__ float hsig(float x) {
    return fminf(fmaxf(0.2f * x + 0.5f, 0.f), 1.f);
}

// ConvLSTM gating: z = (B,h,w,4F) gates i,f,g,o ; updates h,c in place
__global__ void k_lstm_gates(float* h, float* c, const float* z, int F, int n) {
    int i = blockIdx.x * blockDim.x + threadIdx.x;
    if (i >= n) return;
    int p = i / F, ch = i - p * F;
    const float* zp = z + (size_t)p * 4 * F;
    float zi = zp[ch], zf = zp[F + ch], zg = zp[2 * F + ch], zo = zp[3 * F + ch];
    float cn = hsig(zf) * c[i] + hsig(zi) * tanhf(zg);
    c[i] = cn;
    h[i] = hsig(zo) * tanhf(cn);
}

// copy x[:, t] (B,T,H,W,3) -> contiguous (B,HW,3)
__global__ void k_slice_x(float* dst, const float* x, int t, int T, int HW, int C, int n) {
    int i = blockIdx.x * blockDim.x + threadIdx.x;
    if (i >= n) return;
    int c = i % C, p = i / C;
    int s = p % HW, b = p / HW;
    dst[i] = x[((size_t)(b * T + t) * HW + s) * C + c];
}

// a = BN(relu(araw)); [l0: satlu + BN0]; pred[b,t,...] = a; E = [relu(a-A), relu(A-a)]
__global__ void k_ahat_err(float* pred, float* E, const float* araw, const float* A,
                           const float* g, const float* be, const float* mn, const float* vr,
                           const float* b0g, const float* b0b, const float* b0m, const float* b0v,
                           int t, int T, int HW, int C, int n) {
    int i = blockIdx.x * blockDim.x + threadIdx.x;
    if (i >= n) return;
    int c = i % C, p = i / C;
    int s = p % HW, b = p / HW;
    float a = fmaxf(araw[i], 0.f);
    a = (a - mn[c]) * (g[c] * rsqrtf(vr[c] + BN_EPS)) + be[c];
    if (b0g) {
        a = fminf(fmaxf(a, 0.f), 1.f);
        a = (a - b0m[c]) * (b0g[c] * rsqrtf(b0v[c] + BN_EPS)) + b0b[c];
    }
    pred[((size_t)(b * T + t) * HW + s) * C + c] = a;
    float av = A[i];
    E[(size_t)p * 2 * C + c]     = fmaxf(a - av, 0.f);
    E[(size_t)p * 2 * C + C + c] = fmaxf(av - a, 0.f);
}

// 2x2 max pool, stride 2
__global__ void k_maxpool2(float* out, const float* in, int Ho, int Wo, int C, int n) {
    int i = blockIdx.x * blockDim.x + threadIdx.x;
    if (i >= n) return;
    int c = i % C, p = i / C;
    int x = p % Wo; p /= Wo;
    int y = p % Ho; int b = p / Ho;
    int Wi = 2 * Wo;
    size_t base = ((size_t)(b * 2 * Ho + 2 * y) * Wi + 2 * x) * C + c;
    size_t row2 = base + (size_t)Wi * C;
    out[i] = fmaxf(fmaxf(in[base], in[base + C]), fmaxf(in[row2], in[row2 + C]));
}

// =====================================================================
// Host orchestration
// =====================================================================
extern "C" void kernel_launch(void* const* d_in, const int* in_sizes, int n_in,
                              void* d_out, int out_size, void* d_ws, size_t ws_size,
                              hipStream_t stream)
{
    (void)in_sizes; (void)n_in; (void)out_size; (void)ws_size;
    static const int RCH[4]  = {3, 48, 96, 192};
    static const int ACH[4]  = {3, 48, 96, 192};
    static const int INCH[4] = {54, 192, 384, 384};   // 2*A + R[l+1] (top layer: 2*A)
    const int Bn = 16, T = 10;
    const int Hs[4] = {128, 64, 32, 16};
    const int Ws_[4] = {160, 80, 40, 20};
    int HWv[4], PX[4];
    for (int l = 0; l < 4; ++l) { HWv[l] = Hs[l] * Ws_[l]; PX[l] = Bn * HWv[l]; }

    // ---- unpack inputs (setup_inputs dict flattening order) ----
    int ii = 0;
    const float* x = (const float*)d_in[ii++];
    const float *Wx[4], *Wh[4], *lb[4];
    for (int l = 0; l < 4; ++l) {
        Wx[l] = (const float*)d_in[ii++];
        Wh[l] = (const float*)d_in[ii++];
        lb[l] = (const float*)d_in[ii++];
    }
    const float *cW[4], *cb[4], *gg[4], *bb[4], *mm[4], *vv[4];
    for (int l = 0; l < 4; ++l) {
        cW[l] = (const float*)d_in[ii++];
        cb[l] = (const float*)d_in[ii++];
        gg[l] = (const float*)d_in[ii++];
        bb[l] = (const float*)d_in[ii++];
        mm[l] = (const float*)d_in[ii++];
        vv[l] = (const float*)d_in[ii++];
    }
    const float* b0g = (const float*)d_in[ii++];
    const float* b0b = (const float*)d_in[ii++];
    const float* b0m = (const float*)d_in[ii++];
    const float* b0v = (const float*)d_in[ii++];
    const float *uW[3], *ub[3];
    for (int l = 0; l < 3; ++l) {
        uW[l] = (const float*)d_in[ii++];
        ub[l] = (const float*)d_in[ii++];
    }

    // ---- workspace carve (bump allocator, 256B aligned) ----
    char* wp = (char*)d_ws;
    auto alloc_f = [&](size_t nf) -> float* {
        float* p = (float*)wp;
        wp += ((nf * sizeof(float) + 255) / 256) * 256;
        return p;
    };
    auto alloc_h = [&](size_t nh) -> _Float16* {
        _Float16* p = (_Float16*)wp;
        wp += ((nh * sizeof(_Float16) + 255) / 256) * 256;
        return p;
    };

    float *hb[4], *cbuf[4], *Eb[4];
    for (int l = 0; l < 4; ++l) {
        hb[l]   = alloc_f((size_t)PX[l] * RCH[l]);
        cbuf[l] = alloc_f((size_t)PX[l] * RCH[l]);
        Eb[l]   = alloc_f((size_t)PX[l] * 2 * ACH[l]);
    }
    float* inp_s  = alloc_f((size_t)PX[0] * INCH[0]);    // max concat input (l0)
    float* z_s    = alloc_f((size_t)PX[1] * 4 * RCH[1]); // max gate buffer (l1)
    float* araw_s = alloc_f((size_t)PX[1] * ACH[1]);     // max ahat raw (l1)
    float* upt_s  = alloc_f((size_t)PX[0] * ACH[1]);     // max pre-pool conv (l0)
    float* Aping  = alloc_f((size_t)PX[1] * ACH[1]);
    float* Apong  = alloc_f((size_t)PX[1] * ACH[1]);

    // f16 transposed+padded weights: [Npad][Kpad], Npad=ceil64(Cout), Kpad=ceil32(9*Cin)
    auto prep = [&](const float* src, int Cin, int Cout) -> _Float16* {
        int Ktot = 9 * Cin;
        int Kpad = (Ktot + 31) & ~31;
        int Npad = ((Cout + 63) / 64) * 64;
        int n = Npad * Kpad;
        _Float16* d = alloc_h((size_t)n);
        k_wtr<<<(n + 255) / 256, 256, 0, stream>>>(d, src, Cout, Ktot, Kpad, n);
        return d;
    };

    auto zero = [&](float* p, size_t nf) {
        int n = (int)nf;
        k_zero<<<(n + 255) / 256, 256, 0, stream>>>(p, n);
    };
    auto conv = [&](float* o, const float* inb, const _Float16* w, const float* bias,
                    int lh, int lw, int Cin, int Cout, int acc) {
        int M = Bn * lh * lw;
        dim3 g((M + 127) / 128, (Cout + 63) / 64);
        k_conv3x3_wmma<<<g, 256, 0, stream>>>(o, inb, w, bias, Bn, lh, lw, Cin, Cout, acc);
    };

    // ---- per-call init: weight prep + zero recurrent state ----
    _Float16 *fWx[4], *fWh[4], *fcW[4], *fuW[3];
    for (int l = 0; l < 4; ++l) {
        fWx[l] = prep(Wx[l], INCH[l], 4 * RCH[l]);
        fWh[l] = prep(Wh[l], RCH[l], 4 * RCH[l]);
        fcW[l] = prep(cW[l], RCH[l], ACH[l]);
        zero(hb[l], (size_t)PX[l] * RCH[l]);
        zero(cbuf[l], (size_t)PX[l] * RCH[l]);
        zero(Eb[l], (size_t)PX[l] * 2 * ACH[l]);
    }
    for (int l = 0; l < 3; ++l) fuW[l] = prep(uW[l], 2 * ACH[l], ACH[l + 1]);

    // ---- output offsets: preds concatenated per layer, (B,T,h,w,C) ----
    size_t off[4];
    off[0] = 0;
    for (int l = 1; l < 4; ++l)
        off[l] = off[l - 1] + (size_t)Bn * T * HWv[l - 1] * ACH[l - 1];
    float* outp = (float*)d_out;

    // ---- timestep loop ----
    for (int t = 0; t < T; ++t) {
        // top-down: ConvLSTM updates, l = 3..0
        for (int l = 3; l >= 0; --l) {
            const float* inp;
            if (l == 3) {
                inp = Eb[3];  // in_ch == 2*A[3], no concat needed
            } else {
                int nC = PX[l] * 2 * ACH[l];
                k_copy_off<<<(nC + 255) / 256, 256, 0, stream>>>(
                    inp_s, Eb[l], 2 * ACH[l], INCH[l], 0, nC);
                int nU = PX[l] * RCH[l + 1];
                k_upsample_off<<<(nU + 255) / 256, 256, 0, stream>>>(
                    inp_s, hb[l + 1], Hs[l], Ws_[l], RCH[l + 1], INCH[l], 2 * ACH[l], nU);
                inp = inp_s;
            }
            conv(z_s, inp, fWx[l], lb[l], Hs[l], Ws_[l], INCH[l], 4 * RCH[l], 0);
            conv(z_s, hb[l], fWh[l], nullptr, Hs[l], Ws_[l], RCH[l], 4 * RCH[l], 1);
            int nG = PX[l] * RCH[l];
            k_lstm_gates<<<(nG + 255) / 256, 256, 0, stream>>>(hb[l], cbuf[l], z_s, RCH[l], nG);
        }
        // bottom-up: predictions, errors, feedforward
        {
            int n0 = PX[0] * 3;
            k_slice_x<<<(n0 + 255) / 256, 256, 0, stream>>>(Aping, x, t, T, HWv[0], 3, n0);
        }
        float* Acur = Aping;
        float* Anext = Apong;
        for (int l = 0; l < 4; ++l) {
            conv(araw_s, hb[l], fcW[l], cb[l], Hs[l], Ws_[l], RCH[l], ACH[l], 0);
            int nE = PX[l] * ACH[l];
            k_ahat_err<<<(nE + 255) / 256, 256, 0, stream>>>(
                outp + off[l], Eb[l], araw_s, Acur,
                gg[l], bb[l], mm[l], vv[l],
                l == 0 ? b0g : nullptr, l == 0 ? b0b : nullptr,
                l == 0 ? b0m : nullptr, l == 0 ? b0v : nullptr,
                t, T, HWv[l], ACH[l], nE);
            if (l < 3) {
                conv(upt_s, Eb[l], fuW[l], ub[l], Hs[l], Ws_[l], 2 * ACH[l], ACH[l + 1], 0);
                int nP = PX[l + 1] * ACH[l + 1];
                k_maxpool2<<<(nP + 255) / 256, 256, 0, stream>>>(
                    Anext, upt_s, Hs[l + 1], Ws_[l + 1], ACH[l + 1], nP);
                float* tp = Acur; Acur = Anext; Anext = tp;
            }
        }
    }
}
